// MoELayer_13606456394101
// MI455X (gfx1250) — matrix operand order repaired and checked by reading, not compiled, not run on hardware
//
#include <hip/hip_runtime.h>
#include <hip/hip_bf16.h>
#include <math.h>

#define Bb 4
#define Ss 1024
#define Dd 1024
#define Ff 4096
#define Ee 8
#define NTOK (Bb*Ss)            // 4096 tokens
#define OUT_ELEMS (NTOK*Dd)     // 4194304
#define MT 32                   // tokens per block (2 WMMA M-subtiles)
#define TILES_PER_E 128         // 4096/32
#define FC 1024                 // F chunk per phase1/phase2 round
#define NCH (Ff/FC)             // 4 chunks

typedef __attribute__((ext_vector_type(16))) __bf16 v16bf;
typedef __attribute__((ext_vector_type(8)))  float  v8f;

union BFrag {
    v16bf v;
    unsigned short u[16];
    uint4 q[2];
};

// native single-instruction f32 -> bf16 (RNE) via hardware cvt
__device__ __forceinline__ unsigned short f32_to_bf16(float f) {
    __bf16 h = (__bf16)f;
    return __builtin_bit_cast(unsigned short, h);
}

// ---------------- zero init ----------------
__global__ void zero_kernel(float* __restrict__ out, int n,
                            int* __restrict__ counts, float* __restrict__ usage) {
    int i = blockIdx.x * 256 + threadIdx.x;
    if (i < n) out[i] = 0.0f;
    if (i < Ee) { counts[i] = 0; usage[i] = 0.0f; }
}

// ---------------- weight convert + transpose: [E][R][C] f32 -> [E][C][R] bf16 ----
__global__ void convert_w_kernel(const float* __restrict__ w,
                                 unsigned short* __restrict__ wt,
                                 int R, int C) {
    __shared__ float tile[32][33];
    int e  = blockIdx.z;
    int r0 = blockIdx.y * 32;
    int c0 = blockIdx.x * 32;
    int tx = threadIdx.x & 31, ty = threadIdx.x >> 5;   // 32 x 8
    const float* wp = w + (size_t)e * R * C;
#pragma unroll
    for (int i = 0; i < 4; ++i) {
        int r = r0 + ty + i * 8;
        tile[ty + i * 8][tx] = wp[(size_t)r * C + (c0 + tx)];
    }
    __syncthreads();
    unsigned short* op = wt + (size_t)e * R * C;
#pragma unroll
    for (int i = 0; i < 4; ++i) {
        int c = c0 + ty + i * 8;
        op[(size_t)c * R + (r0 + tx)] = f32_to_bf16(tile[tx][ty + i * 8]);
    }
}

// ---------------- gating: one wave32 per token ----------------
__global__ void gate_kernel(const float* __restrict__ x,
                            const float* __restrict__ gw,
                            const float* __restrict__ gb,
                            int* __restrict__ counts,
                            int* __restrict__ toks,
                            float* __restrict__ wts,
                            float* __restrict__ usage) {
    __shared__ float susage[Ee];
    int tid  = threadIdx.x;
    if (tid < Ee) susage[tid] = 0.0f;
    __syncthreads();

    int wave = tid >> 5, lane = tid & 31;
    int token = blockIdx.x * 8 + wave;

    float acc[Ee] = {0,0,0,0,0,0,0,0};
    const float* xp = x + (size_t)token * Dd;
    for (int d = lane; d < Dd; d += 32) {
        float xv = xp[d];
        const float4* g = (const float4*)(gw + (size_t)d * Ee);
        float4 g0 = g[0], g1 = g[1];
        acc[0] += xv * g0.x; acc[1] += xv * g0.y;
        acc[2] += xv * g0.z; acc[3] += xv * g0.w;
        acc[4] += xv * g1.x; acc[5] += xv * g1.y;
        acc[6] += xv * g1.z; acc[7] += xv * g1.w;
    }
#pragma unroll
    for (int e = 0; e < Ee; ++e)
#pragma unroll
        for (int off = 16; off; off >>= 1)
            acc[e] += __shfl_xor(acc[e], off, 32);

    if (lane == 0) {
        float l[Ee];
#pragma unroll
        for (int e = 0; e < Ee; ++e) l[e] = acc[e] + gb[e];
        float m = l[0];
#pragma unroll
        for (int e = 1; e < Ee; ++e) m = fmaxf(m, l[e]);
        float p[Ee], s = 0.0f;
#pragma unroll
        for (int e = 0; e < Ee; ++e) { p[e] = __expf(l[e] - m); s += p[e]; }
        float inv = 1.0f / s;
#pragma unroll
        for (int e = 0; e < Ee; ++e) atomicAdd(&susage[e], p[e] * inv);
        int i0 = 0;
#pragma unroll
        for (int e = 1; e < Ee; ++e) if (l[e] > l[i0]) i0 = e;
        int i1 = (i0 == 0) ? 1 : 0;
#pragma unroll
        for (int e = 0; e < Ee; ++e) if (e != i0 && l[e] > l[i1]) i1 = e;
        float p0 = 1.0f / (1.0f + __expf(l[i1] - l[i0]));
        float p1 = 1.0f - p0;
        int pos0 = atomicAdd(&counts[i0], 1);
        toks[i0 * NTOK + pos0] = token; wts[i0 * NTOK + pos0] = p0;
        int pos1 = atomicAdd(&counts[i1], 1);
        toks[i1 * NTOK + pos1] = token; wts[i1 * NTOK + pos1] = p1;
    }
    __syncthreads();
    if (tid < Ee) unsafeAtomicAdd(&usage[tid], susage[tid]);
}

// ---------------- expert FFN: 32-token tile per block, fused F-chunked FFN ----------------
// LDS: Abuf 32x1024 bf16 (64KB) | Hbuf 32x1024 bf16 (64KB) | meta
#define LDS_ABUF 0
#define LDS_HBUF 65536
#define LDS_TOK  131072
#define LDS_CW   131200
#define LDS_VAL  131328
#define LDS_TOTAL 131456

__global__ void ffn_kernel(const float* __restrict__ x,
                           const unsigned short* __restrict__ w1T, // [E][F][D] bf16
                           const float* __restrict__ b1,
                           const unsigned short* __restrict__ w2T, // [E][D][F] bf16
                           const float* __restrict__ b2,
                           const int*   __restrict__ counts,
                           const int*   __restrict__ toks,
                           const float* __restrict__ wts,
                           float* __restrict__ out) {
    extern __shared__ char smem[];
    unsigned short* Abuf = (unsigned short*)(smem + LDS_ABUF);
    unsigned short* Hbuf = (unsigned short*)(smem + LDS_HBUF);
    int*   sTok   = (int*)  (smem + LDS_TOK);
    float* sCw    = (float*)(smem + LDS_CW);
    int*   sValid = (int*)  (smem + LDS_VAL);

    int e    = blockIdx.x >> 7;              // 128 tiles per expert
    int tile = blockIdx.x & (TILES_PER_E - 1);
    int count = counts[e];
    if (tile * MT >= count) return;          // uniform block exit

    int tid = threadIdx.x;
    if (tid < MT) {
        int slot = tile * MT + tid;
        if (slot < count) {
            sTok[tid] = toks[e * NTOK + slot];
            sCw[tid]  = wts[e * NTOK + slot];
            sValid[tid] = 1;
        } else {
            sTok[tid] = 0; sCw[tid] = 0.0f; sValid[tid] = 0;
        }
    }
    __syncthreads();

    // stage gathered x rows as bf16 into LDS (32 x 1024)
    for (int i = tid; i < MT * Dd; i += 256) {
        int r = i >> 10, c = i & (Dd - 1);
        Abuf[i] = f32_to_bf16(x[(size_t)sTok[r] * Dd + c]);
    }
    __syncthreads();

    int wave = tid >> 5, lane = tid & 31;
    int MN     = lane & 15;                   // A row / B col within tile
    int c0     = (lane < 16) ? 0 : 8;         // A-frag K split
    int koff   = (lane < 16) ? 0 : 8;         // B-frag K split
    int mshift = (lane < 16) ? 0 : 8;         // C/D row split

    const unsigned short* w1t = w1T + ((size_t)e * Ff * Dd);
    const unsigned short* w2t = w2T + ((size_t)e * Dd * Ff);

    // persistent phase-2 accumulators: 2 M-subtiles x 8 N-tiles (128 D cols/wave)
    v8f acc2[2][8];
#pragma unroll
    for (int m = 0; m < 2; ++m)
#pragma unroll
        for (int i = 0; i < 8; ++i) acc2[m][i] = (v8f){};

    // one base pointer; all 8 phase-2 N-tiles are reached by constant offsets i*16*Ff
    const unsigned short* b2base = w2t + (size_t)(wave * 128 + MN) * Ff + koff;

#pragma unroll 1
    for (int ch = 0; ch < NCH; ++ch) {
        int fbase = ch * FC;

        // ---- phase 1: h[:, chunk] = gelu(x @ w1[:, chunk] + b1); wave owns 128 F-cols ----
#pragma unroll 1
        for (int ng = 0; ng < 8; ng += 4) {
            v8f acc[2][4];
#pragma unroll
            for (int m = 0; m < 2; ++m)
#pragma unroll
                for (int i = 0; i < 4; ++i) acc[m][i] = (v8f){};
            // base for this N-group; tiles at constant offsets i*16*Dd
            const unsigned short* bbase =
                w1t + (size_t)(fbase + wave * 128 + ng * 16 + MN) * Dd + koff;
#pragma unroll 1
            for (int k0 = 0; k0 < Dd; k0 += 32) {
                BFrag A0, A1, Bf[4];
                A0.q[0] = *(const uint4*)&Abuf[MN * Dd + k0 + c0];
                A0.q[1] = *(const uint4*)&Abuf[MN * Dd + k0 + c0 + 16];
                A1.q[0] = *(const uint4*)&Abuf[(16 + MN) * Dd + k0 + c0];
                A1.q[1] = *(const uint4*)&Abuf[(16 + MN) * Dd + k0 + c0 + 16];
                __builtin_prefetch(bbase + k0 + 256, 0, 1);
#pragma unroll
                for (int i = 0; i < 4; ++i) {
                    const unsigned short* bp = bbase + (size_t)i * 16 * Dd + k0;
                    Bf[i].q[0] = *(const uint4*)(bp);
                    Bf[i].q[1] = *(const uint4*)(bp + 16);
                }
#pragma unroll
                for (int i = 0; i < 4; ++i) {
                    acc[0][i] = __builtin_amdgcn_wmma_f32_16x16x32_bf16(
                        false, A0.v, false, Bf[i].v, (short)0, acc[0][i], false, false);
                    acc[1][i] = __builtin_amdgcn_wmma_f32_16x16x32_bf16(
                        false, A1.v, false, Bf[i].v, (short)0, acc[1][i], false, false);
                }
            }
#pragma unroll
            for (int i = 0; i < 4; ++i) {
                int nc = wave * 128 + (ng + i) * 16 + MN;        // col within chunk
                float bias = b1[e * Ff + fbase + nc];
#pragma unroll
                for (int m = 0; m < 2; ++m)
#pragma unroll
                    for (int r = 0; r < 8; ++r) {
                        float v = acc[m][i][r] + bias;
                        v = 0.5f * v * (1.0f + erff(v * 0.70710678118654752f));
                        Hbuf[(m * 16 + r + mshift) * FC + nc] = f32_to_bf16(v);
                    }
            }
        }
        __syncthreads();

        // ---- phase 2 partial: acc2 += h[:, chunk] @ w2[chunk, :] ----
#pragma unroll 1
        for (int k0 = 0; k0 < FC; k0 += 32) {
            BFrag A0, A1;
            A0.q[0] = *(const uint4*)&Hbuf[MN * FC + k0 + c0];
            A0.q[1] = *(const uint4*)&Hbuf[MN * FC + k0 + c0 + 16];
            A1.q[0] = *(const uint4*)&Hbuf[(16 + MN) * FC + k0 + c0];
            A1.q[1] = *(const uint4*)&Hbuf[(16 + MN) * FC + k0 + c0 + 16];
            const unsigned short* kb = b2base + fbase + k0;
            __builtin_prefetch(kb + 256, 0, 1);
#pragma unroll
            for (int g = 0; g < 2; ++g) {
                BFrag Bf[4];
#pragma unroll
                for (int i = 0; i < 4; ++i) {
                    const unsigned short* bp = kb + (size_t)(g * 4 + i) * 16 * Ff;
                    Bf[i].q[0] = *(const uint4*)(bp);
                    Bf[i].q[1] = *(const uint4*)(bp + 16);
                }
#pragma unroll
                for (int i = 0; i < 4; ++i) {
                    acc2[0][g * 4 + i] = __builtin_amdgcn_wmma_f32_16x16x32_bf16(
                        false, A0.v, false, Bf[i].v, (short)0, acc2[0][g * 4 + i], false, false);
                    acc2[1][g * 4 + i] = __builtin_amdgcn_wmma_f32_16x16x32_bf16(
                        false, A1.v, false, Bf[i].v, (short)0, acc2[1][g * 4 + i], false, false);
                }
            }
        }
        __syncthreads();   // Hbuf reused by next chunk's phase 1
    }

    // ---- epilogue: out += cw * (acc2 + b2) ----
#pragma unroll
    for (int i = 0; i < 8; ++i) {
        int n0 = wave * 128 + i * 16;
        float bias = b2[e * Dd + n0 + MN];
#pragma unroll
        for (int m = 0; m < 2; ++m)
#pragma unroll
            for (int r = 0; r < 8; ++r) {
                int row = m * 16 + r + mshift;
                if (sValid[row]) {
                    float v = (acc2[m][i][r] + bias) * sCw[row];
                    unsafeAtomicAdd(&out[(size_t)sTok[row] * Dd + (n0 + MN)], v);
                }
            }
    }
}

// ---------------- aux loss ----------------
__global__ void finalize_kernel(const float* __restrict__ usage, float* __restrict__ auxp) {
    if (threadIdx.x == 0 && blockIdx.x == 0) {
        float s = 0.0f;
        for (int e = 0; e < Ee; ++e) {
            float u = usage[e] * (1.0f / (float)NTOK);
            s += u * u;
        }
        *auxp = (float)Ee * s;
    }
}

extern "C" void kernel_launch(void* const* d_in, const int* in_sizes, int n_in,
                              void* d_out, int out_size, void* d_ws, size_t ws_size,
                              hipStream_t stream) {
    const float* x      = (const float*)d_in[0];
    const float* gate_w = (const float*)d_in[1];
    const float* gate_b = (const float*)d_in[2];
    const float* w1     = (const float*)d_in[3];
    const float* b1     = (const float*)d_in[4];
    const float* w2     = (const float*)d_in[5];
    const float* b2     = (const float*)d_in[6];
    float* out = (float*)d_out;

    // workspace layout
    char* ws = (char*)d_ws;
    int*   counts = (int*)  (ws + 0);                     // 8 ints
    float* usage  = (float*)(ws + 128);                   // 8 floats
    int*   toks   = (int*)  (ws + 256);                   // 8*4096 ints
    float* wts    = (float*)(ws + 256 + Ee * NTOK * 4);   // 8*4096 floats
    unsigned short* w1T = (unsigned short*)(ws + 524288);               // 64 MB
    unsigned short* w2T = (unsigned short*)(ws + 524288 + (size_t)Ee*Ff*Dd*2); // 64 MB

    int zeroN = OUT_ELEMS + 1;
    zero_kernel<<<(zeroN + 255) / 256, 256, 0, stream>>>(out, zeroN, counts, usage);

    // w1: [E][D][F] -> w1T [E][F][D] ; w2: [E][F][D] -> w2T [E][D][F]
    convert_w_kernel<<<dim3(Ff / 32, Dd / 32, Ee), 256, 0, stream>>>(w1, w1T, Dd, Ff);
    convert_w_kernel<<<dim3(Dd / 32, Ff / 32, Ee), 256, 0, stream>>>(w2, w2T, Ff, Dd);

    gate_kernel<<<NTOK / 8, 256, 0, stream>>>(x, gate_w, gate_b, counts, toks, wts, usage);

    ffn_kernel<<<Ee * TILES_PER_E, 256, LDS_TOTAL, stream>>>(
        x, w1T, b1, w2T, b2, counts, toks, wts, out);

    finalize_kernel<<<1, 32, 0, stream>>>(usage, out + OUT_ELEMS);
}
